// FeedForward_40492951667103
// MI455X (gfx1250) — matrix operand order repaired and checked by reading, not compiled, not run on hardware
//
#include <hip/hip_runtime.h>

typedef __bf16 bf16;
typedef __attribute__((ext_vector_type(16))) __bf16 v16bf;
typedef __attribute__((ext_vector_type(8)))  float  v8f;

#define T_TOK 4096
#define DIM   1024
#define FDIM  768
#define NEXP  64
#define TOPK  8
#define BM    64
#define BN    128
#define BK    64
// max padded rows: 32768 + 64*(BM-1) = 36800 ; tiles = 36800/64 = 575
#define MAX_ROWS  36800
#define MAX_TILES 575

// ---------------- gate: logits + softmax + top-8 + renorm --------------------
__global__ void __launch_bounds__(256)
gate_topk_kernel(const float* __restrict__ x, const float* __restrict__ gw,
                 int* __restrict__ topk_ids, float* __restrict__ topk_w,
                 int* __restrict__ cnt) {
  __shared__ float xrow[DIM];
  __shared__ float sc[NEXP];
  const int t = blockIdx.x;
  const float* xr = x + (size_t)t * DIM;
  for (int i = threadIdx.x; i < DIM; i += blockDim.x) xrow[i] = xr[i];
  __syncthreads();
  if (threadIdx.x < NEXP) {
    const float* w = gw + (size_t)threadIdx.x * DIM;
    float acc = 0.f;
    for (int i = 0; i < DIM; i += 4)
      acc += xrow[i]*w[i] + xrow[i+1]*w[i+1] + xrow[i+2]*w[i+2] + xrow[i+3]*w[i+3];
    sc[threadIdx.x] = acc;
  }
  __syncthreads();
  if (threadIdx.x == 0) {
    float mx = sc[0];
    for (int e = 1; e < NEXP; e++) mx = fmaxf(mx, sc[e]);
    float sum = 0.f;
    for (int e = 0; e < NEXP; e++) { sc[e] = __expf(sc[e] - mx); sum += sc[e]; }
    const float inv = 1.f / sum;
    int   ids[TOPK]; float wv[TOPK]; float wsum = 0.f;
    for (int k = 0; k < TOPK; k++) {
      int best = 0; float bv = -1.f;
      for (int e = 0; e < NEXP; e++) if (sc[e] > bv) { bv = sc[e]; best = e; }
      ids[k] = best; wv[k] = bv * inv; wsum += bv * inv; sc[best] = -1.f;
    }
    const float r = 1.f / (wsum + 1e-20f);
    for (int k = 0; k < TOPK; k++) {
      topk_ids[t*TOPK + k] = ids[k];
      topk_w [t*TOPK + k] = wv[k] * r;
      atomicAdd(&cnt[ids[k]], 1);
    }
  }
}

// ---------------- routing: padded offsets + tile->expert table ---------------
__global__ void routing_kernel(const int* __restrict__ cnt, int* __restrict__ base,
                               int* __restrict__ tile_expert) {
  if (threadIdx.x == 0) {
    int off = 0;
    for (int e = 0; e < NEXP; e++) {
      base[e] = off;
      int pad = (cnt[e] + BM - 1) / BM * BM;
      int t0 = off / BM, t1 = (off + pad) / BM;
      for (int ti = t0; ti < t1; ti++) tile_expert[ti] = e;
      off += pad;
    }
  }
}

// ---------------- scatter: gather token rows (fp32 -> bf16) ------------------
__global__ void __launch_bounds__(256)
scatter_kernel(const float* __restrict__ x, const int* __restrict__ topk_ids,
               const int* __restrict__ base, int* __restrict__ fill,
               int* __restrict__ row_map, bf16* __restrict__ xs) {
  __shared__ int srow;
  const int f = blockIdx.x;          // flat expanded index = t*TOPK + k
  const int t = f >> 3;
  if (threadIdx.x == 0) {
    int e = topk_ids[f];
    int slot = atomicAdd(&fill[e], 1);
    int row = base[e] + slot;
    row_map[row] = f;
    srow = row;
  }
  __syncthreads();
  const int row = srow;
  const float* xr = x + (size_t)t * DIM;
  bf16* dst = xs + (size_t)row * DIM;
  for (int i = threadIdx.x; i < DIM; i += blockDim.x) dst[i] = (bf16)xr[i];
}

__device__ __forceinline__ float fast_silu(float g) {
  // g * sigmoid(g) with v_rcp_f32 instead of IEEE division
  return g * __builtin_amdgcn_rcpf(1.f + __expf(-g));
}

// ---------------- GEMM1: fused gate/up + SwiGLU (bf16 WMMA) ------------------
// BM=64 x BN=128 per workgroup; 8 waves in 2x4; each wave: 2x2 subtiles of 16x16.
__global__ void __launch_bounds__(256)
gemm1_kernel(const bf16* __restrict__ xs, const float* __restrict__ w_gate,
             const float* __restrict__ w_up, const int* __restrict__ tile_expert,
             bf16* __restrict__ h) {
  const int e = tile_expert[blockIdx.x];
  if (e < 0) return;
  const int m0 = blockIdx.x * BM;
  const int n0 = blockIdx.y * BN;
  __shared__ bf16 As[BM][BK];   // 8 KB
  __shared__ bf16 Bg[BN][BK];   // 16 KB
  __shared__ bf16 Bu[BN][BK];   // 16 KB
  const int tid = threadIdx.x;
  const int lane = tid & 31;
  const int wave = tid >> 5;
  const int wy = wave >> 2, wx = wave & 3;  // 2 x 4 waves
  const int hi = lane >> 4;
  const int l15 = lane & 15;
  v8f accg[2][2] = {}; v8f accu[2][2] = {};
  const float* wg = w_gate + ((size_t)e * FDIM + n0) * DIM;
  const float* wu = w_up   + ((size_t)e * FDIM + n0) * DIM;

  for (int k0 = 0; k0 < DIM; k0 += BK) {
    { // A tile: 64x64 bf16 = 4096 elems, 16/thread (32B contiguous)
      int row = tid >> 2, c = (tid & 3) << 4;
      const bf16* src = xs + (size_t)(m0 + row) * DIM + k0 + c;
      #pragma unroll
      for (int j = 0; j < 16; j++) As[row][c + j] = src[j];
    }
    { // B tiles: 128x64 f32 -> bf16 = 8192 elems, 32/thread each
      int row = tid >> 1, c = (tid & 1) << 5;
      const float* sg = wg + (size_t)row * DIM + k0 + c;
      const float* su = wu + (size_t)row * DIM + k0 + c;
      if (k0 + BK < DIM) { __builtin_prefetch(sg + BK, 0, 1); __builtin_prefetch(su + BK, 0, 1); }
      #pragma unroll
      for (int j = 0; j < 32; j++) Bg[row][c + j] = (bf16)sg[j];
      #pragma unroll
      for (int j = 0; j < 32; j++) Bu[row][c + j] = (bf16)su[j];
    }
    __syncthreads();
    #pragma unroll
    for (int kk = 0; kk < BK; kk += 32) {
      const int ka = kk + (hi << 4);   // A: lanes 0-15 -> K 0..15 ; 16-31 -> 16..31
      const int kb = kk + (hi << 4);   // B: lanes 0-15 -> K 0..15 ; 16-31 -> 16..31
      v16bf a[2], bg[2], bu[2];
      #pragma unroll
      for (int mi = 0; mi < 2; mi++) {
        const int m = wy * 32 + mi * 16 + l15;
        #pragma unroll
        for (int j = 0; j < 8; j++) { a[mi][j] = As[m][ka + j]; a[mi][j + 8] = As[m][ka + 8 + j]; }
      }
      #pragma unroll
      for (int ni = 0; ni < 2; ni++) {
        const int n = wx * 32 + ni * 16 + l15;
        #pragma unroll
        for (int j = 0; j < 16; j++) { bg[ni][j] = Bg[n][kb + j]; bu[ni][j] = Bu[n][kb + j]; }
      }
      #pragma unroll
      for (int mi = 0; mi < 2; mi++)
        #pragma unroll
        for (int ni = 0; ni < 2; ni++) {
          accg[mi][ni] = __builtin_amdgcn_wmma_f32_16x16x32_bf16(false, a[mi], false, bg[ni], (short)0, accg[mi][ni], false, false);
          accu[mi][ni] = __builtin_amdgcn_wmma_f32_16x16x32_bf16(false, a[mi], false, bu[ni], (short)0, accu[mi][ni], false, false);
        }
    }
    __syncthreads();
  }
  // epilogue: h = silu(g) * u
  #pragma unroll
  for (int mi = 0; mi < 2; mi++) {
    const int rbase = m0 + wy * 32 + mi * 16 + (hi << 3);
    #pragma unroll
    for (int ni = 0; ni < 2; ni++) {
      const int col = n0 + wx * 32 + ni * 16 + l15;
      #pragma unroll
      for (int r = 0; r < 8; r++) {
        float g = accg[mi][ni][r], u = accu[mi][ni][r];
        h[(size_t)(rbase + r) * FDIM + col] = (bf16)(fast_silu(g) * u);
      }
    }
  }
}

// ---------------- GEMM2: down projection, scatter to flat slots --------------
__global__ void __launch_bounds__(256)
gemm2_kernel(const bf16* __restrict__ h, const float* __restrict__ w_down,
             const int* __restrict__ tile_expert, const int* __restrict__ row_map,
             float* __restrict__ out_exp) {
  const int e = tile_expert[blockIdx.x];
  if (e < 0) return;
  const int m0 = blockIdx.x * BM;
  const int n0 = blockIdx.y * BN;
  __shared__ bf16 As[BM][BK];   // 8 KB
  __shared__ bf16 Bd[BN][BK];   // 16 KB
  const int tid = threadIdx.x;
  const int lane = tid & 31;
  const int wave = tid >> 5;
  const int wy = wave >> 2, wx = wave & 3;
  const int hi = lane >> 4;
  const int l15 = lane & 15;
  v8f acc[2][2] = {};
  const float* wd = w_down + ((size_t)e * DIM + n0) * FDIM;

  for (int k0 = 0; k0 < FDIM; k0 += BK) {
    {
      int row = tid >> 2, c = (tid & 3) << 4;
      const bf16* src = h + (size_t)(m0 + row) * FDIM + k0 + c;
      #pragma unroll
      for (int j = 0; j < 16; j++) As[row][c + j] = src[j];
    }
    {
      int row = tid >> 1, c = (tid & 1) << 5;
      const float* sd = wd + (size_t)row * FDIM + k0 + c;
      if (k0 + BK < FDIM) __builtin_prefetch(sd + BK, 0, 1);
      #pragma unroll
      for (int j = 0; j < 32; j++) Bd[row][c + j] = (bf16)sd[j];
    }
    __syncthreads();
    #pragma unroll
    for (int kk = 0; kk < BK; kk += 32) {
      const int ka = kk + (hi << 4);
      const int kb = kk + (hi << 4);
      v16bf a[2], b[2];
      #pragma unroll
      for (int mi = 0; mi < 2; mi++) {
        const int m = wy * 32 + mi * 16 + l15;
        #pragma unroll
        for (int j = 0; j < 8; j++) { a[mi][j] = As[m][ka + j]; a[mi][j + 8] = As[m][ka + 8 + j]; }
      }
      #pragma unroll
      for (int ni = 0; ni < 2; ni++) {
        const int n = wx * 32 + ni * 16 + l15;
        #pragma unroll
        for (int j = 0; j < 16; j++) b[ni][j] = Bd[n][kb + j];
      }
      #pragma unroll
      for (int mi = 0; mi < 2; mi++)
        #pragma unroll
        for (int ni = 0; ni < 2; ni++)
          acc[mi][ni] = __builtin_amdgcn_wmma_f32_16x16x32_bf16(false, a[mi], false, b[ni], (short)0, acc[mi][ni], false, false);
    }
    __syncthreads();
  }
  #pragma unroll
  for (int mi = 0; mi < 2; mi++) {
    const int rbase = m0 + wy * 32 + mi * 16 + (hi << 3);
    #pragma unroll
    for (int r = 0; r < 8; r++) {
      const int fl = row_map[rbase + r];
      if (fl >= 0) {
        #pragma unroll
        for (int ni = 0; ni < 2; ni++) {
          const int col = n0 + wx * 32 + ni * 16 + l15;
          out_exp[(size_t)fl * DIM + col] = acc[mi][ni][r];
        }
      }
    }
  }
}

// ---------------- combine: weighted sum over the 8 slots per token -----------
__global__ void __launch_bounds__(256)
combine_kernel(const float* __restrict__ out_exp, const float* __restrict__ topk_w,
               float* __restrict__ y) {
  const int t = blockIdx.x;
  for (int d = threadIdx.x; d < DIM; d += blockDim.x) {
    float acc = 0.f;
    #pragma unroll
    for (int k = 0; k < TOPK; k++)
      acc += topk_w[t*TOPK + k] * out_exp[(size_t)(t*TOPK + k) * DIM + d];
    y[(size_t)t * DIM + d] = acc;
  }
}

extern "C" void kernel_launch(void* const* d_in, const int* in_sizes, int n_in,
                              void* d_out, int out_size, void* d_ws, size_t ws_size,
                              hipStream_t stream) {
  (void)in_sizes; (void)n_in; (void)out_size; (void)ws_size;
  const float* x      = (const float*)d_in[0];
  const float* gw     = (const float*)d_in[1];
  const float* w_gate = (const float*)d_in[2];
  const float* w_up   = (const float*)d_in[3];
  const float* w_down = (const float*)d_in[4];
  float* y = (float*)d_out;

  char* ws = (char*)d_ws;
  size_t off = 0;
  auto alloc = [&](size_t bytes) -> void* {
    void* p = ws + off;
    off += (bytes + 255) & ~(size_t)255;
    return p;
  };
  int*   cnt         = (int*)  alloc(NEXP * 4);
  int*   fill        = (int*)  alloc(NEXP * 4);
  int*   base        = (int*)  alloc(NEXP * 4);
  int*   tile_expert = (int*)  alloc(MAX_TILES * 4);
  int*   topk_ids    = (int*)  alloc((size_t)T_TOK * TOPK * 4);
  float* topk_w      = (float*)alloc((size_t)T_TOK * TOPK * 4);
  int*   row_map     = (int*)  alloc((size_t)MAX_ROWS * 4);
  bf16*  xs          = (bf16*) alloc((size_t)MAX_ROWS * DIM * 2);
  bf16*  hbuf        = (bf16*) alloc((size_t)MAX_ROWS * FDIM * 2);
  float* out_exp     = (float*)alloc((size_t)T_TOK * TOPK * DIM * 4);

  hipMemsetAsync(cnt,  0, NEXP * 4, stream);
  hipMemsetAsync(fill, 0, NEXP * 4, stream);
  hipMemsetAsync(tile_expert, 0xFF, MAX_TILES * 4, stream);
  hipMemsetAsync(row_map, 0xFF, (size_t)MAX_ROWS * 4, stream);
  hipMemsetAsync(xs, 0, (size_t)MAX_ROWS * DIM * 2, stream);

  gate_topk_kernel<<<T_TOK, 256, 0, stream>>>(x, gw, topk_ids, topk_w, cnt);
  routing_kernel  <<<1, 64, 0, stream>>>(cnt, base, tile_expert);
  scatter_kernel  <<<T_TOK * TOPK, 256, 0, stream>>>(x, topk_ids, base, fill, row_map, xs);
  gemm1_kernel    <<<dim3(MAX_TILES, FDIM / BN), 256, 0, stream>>>(xs, w_gate, w_up, tile_expert, hbuf);
  gemm2_kernel    <<<dim3(MAX_TILES, DIM / BN), 256, 0, stream>>>(hbuf, w_down, tile_expert, row_map, out_exp);
  combine_kernel  <<<T_TOK, 256, 0, stream>>>(out_exp, topk_w, y);
}
